// CapsNet_1133871366149
// MI455X (gfx1250) — compile-verified, matrix operand order checked by reading, not run on hardware
//
#include <hip/hip_runtime.h>
#include <hip/hip_bf16.h>

// ---------------------------------------------------------------------------
// CapsNet forward for MI455X (gfx1250, wave32, WMMA + async-to-LDS).
//  - conv1: implicit GEMM on v_wmma_f32_16x16x32_f16, emits h in NHWC f16
//  - primary-caps conv: implicit GEMM, K ordered (ky,kx,ci) so B-chunks are
//    contiguous; B tile staged with global_load_async_to_lds_b128, double
//    buffered against the WMMA pipe (ASYNCcnt)
//  - routing: u recomputed on the fly from L2-resident W (never materialized)
//  - decoder: 3 WMMA GEMMs with fused bias+relu/sigmoid
// ---------------------------------------------------------------------------

typedef __attribute__((ext_vector_type(16))) _Float16 v16h;
typedef __attribute__((ext_vector_type(8)))  _Float16 v8h;
typedef __attribute__((ext_vector_type(8)))  float    v8f;

#define B_SZ   512
#define NCAPS  1152
#define JCAPS  10
#define DVEC   16

// ---- WMMA fragment load: row-major f16 matrix, 16 rows x 32 K -------------
// 16-bit A/B layout (ISA 7.12.2): lane L -> row = L%16; halves 0..7 hold
// K = (L>=16?8:0)+e, halves 8..15 hold K = 16+(L>=16?8:0)+(e-8).
__device__ inline v16h load_frag_rm(const _Float16* __restrict__ base, int ld,
                                    int row, int k0, int lane) {
  int khi = (lane >= 16) ? 8 : 0;
  const _Float16* p0 = base + (size_t)row * ld + k0 + khi;
  v8h r1 = *(const v8h*)(p0);
  v8h r2 = *(const v8h*)(p0 + 16);
  v16h out;
#pragma unroll
  for (int e = 0; e < 8; ++e) { out[e] = r1[e]; out[8 + e] = r2[e]; }
  return out;
}

// ---- small utility kernels -------------------------------------------------
__global__ void k_fill_zero_f32(float* p, int n) {
  for (int i = blockIdx.x * blockDim.x + threadIdx.x; i < n; i += gridDim.x * blockDim.x)
    p[i] = 0.f;
}

__global__ void k_cvt_f32_f16(const float* __restrict__ in, _Float16* __restrict__ out, int n) {
  for (int i = blockIdx.x * blockDim.x + threadIdx.x; i < n; i += gridDim.x * blockDim.x)
    out[i] = (_Float16)in[i];
}

// conv1 weights [256,1,9,9] -> f16 [256,96] (K padded 81->96 with zeros)
__global__ void k_pack_conv1_w(const float* __restrict__ in, _Float16* __restrict__ out) {
  int idx = blockIdx.x * blockDim.x + threadIdx.x;
  if (idx >= 256 * 96) return;
  int co = idx / 96, k = idx % 96;
  out[idx] = (k < 81) ? (_Float16)in[co * 81 + k] : (_Float16)0.f;
}

// pc_w [co][ci][ky][kx] f32 -> f16 [co][pix=ky*9+kx][ci]  (K-order (ky,kx,ci))
__global__ void k_pack_pcw(const float* __restrict__ in, _Float16* __restrict__ out) {
  int idx = blockIdx.x * blockDim.x + threadIdx.x;
  if (idx >= 256 * 20736) return;
  int co = idx / 20736, r = idx % 20736;
  int pix = r >> 8, ci = r & 255;
  out[idx] = (_Float16)in[(size_t)co * 20736 + ci * 81 + pix];
}

// f32 [Kin,N] -> f16 [N,Kpad] (transpose + zero-pad K)
__global__ void k_transpose_pad(const float* __restrict__ in, _Float16* __restrict__ out,
                                int Kin, int Kpad, int N) {
  int idx = blockIdx.x * blockDim.x + threadIdx.x;
  int total = N * Kpad;
  if (idx >= total) return;
  int n = idx / Kpad, k = idx % Kpad;
  out[idx] = (k < Kin) ? (_Float16)in[(size_t)k * N + n] : (_Float16)0.f;
}

// ---- conv1: implicit GEMM (Cin=1, 9x9 s1, 28x28 -> 20x20), relu ------------
// GEMM: M=256 (Cout), K=81 pad 96, N=512*400. Output h NHWC f16 [b,20,20,256].
__global__ __launch_bounds__(256) void k_conv1_wmma(const _Float16* __restrict__ w,
                                                    const _Float16* __restrict__ src,
                                                    const float* __restrict__ bias,
                                                    _Float16* __restrict__ h) {
  constexpr int KTOT = 96, KREAL = 81, NTOT = 512 * 400;
  __shared__ _Float16 __attribute__((aligned(32))) blds[32 * 16];

  const int nBase = blockIdx.x * 16;
  const int wave = threadIdx.x >> 5;
  const int lane = threadIdx.x & 31;

  v8f acc0 = {}; v8f acc1 = {};

  for (int kc = 0; kc < KTOT; kc += 32) {
    { // gather B chunk directly in fragment layout
      int idx = threadIdx.x * 2;
      int L = idx >> 4, e0 = idx & 15;
      int col = L & 15, khi = (L >= 16) ? 8 : 0;
      int gn = nBase + col;
      int b_ = gn / 400, rem = gn % 400, oy = rem / 20, ox = rem % 20;
      _Float16 vals[2];
#pragma unroll
      for (int q = 0; q < 2; ++q) {
        int e = e0 + q;
        int K = (e < 8) ? (khi + e) : (16 + khi + (e - 8));
        int gk = kc + K;
        _Float16 vv = (_Float16)0.f;
        if (gn < NTOT && gk < KREAL) {
          int ky = gk / 9, kx = gk % 9;
          vv = src[((size_t)b_ * 28 + (oy + ky)) * 28 + (ox + kx)];
        }
        vals[q] = vv;
      }
      blds[idx] = vals[0];
      blds[idx + 1] = vals[1];
    }
    __syncthreads();
    v16h bf = *(const v16h*)(&blds[lane * 16]);
    int co0 = wave * 32;
    v16h a0 = load_frag_rm(w, KTOT, co0 + (lane & 15), kc, lane);
    v16h a1 = load_frag_rm(w, KTOT, co0 + 16 + (lane & 15), kc, lane);
    acc0 = __builtin_amdgcn_wmma_f32_16x16x32_f16(false, a0, false, bf, (short)0, acc0, false, false);
    acc1 = __builtin_amdgcn_wmma_f32_16x16x32_f16(false, a1, false, bf, (short)0, acc1, false, false);
    __syncthreads();
  }

  int gn = nBase + (lane & 15);
  int b_ = gn / 400, rem = gn % 400, oy = rem / 20, ox = rem % 20;
  size_t spat = (((size_t)b_ * 20 + oy) * 20 + ox) * 256;
  int rowOff = (lane >= 16) ? 8 : 0;
#pragma unroll
  for (int r = 0; r < 8; ++r) {
#pragma unroll
    for (int t = 0; t < 2; ++t) {
      int co = wave * 32 + t * 16 + rowOff + r;
      float val = (t == 0 ? acc0[r] : acc1[r]) + bias[co];
      val = val > 0.f ? val : 0.f;
      h[spat + co] = (_Float16)val;  // NHWC
    }
  }
}

// ---- primary caps conv: implicit GEMM with async double-buffered B --------
// GEMM: M=256, K=20736 ordered (ky,kx,ci), N=512*36. src = h NHWC f16.
// Output p f32 [n,1152,8] (caps layout fused into store).
__global__ __launch_bounds__(256) void k_pcconv_wmma(const _Float16* __restrict__ w,
                                                     const _Float16* __restrict__ src,
                                                     const float* __restrict__ bias,
                                                     float* __restrict__ p) {
  constexpr int KTOT = 20736;
  constexpr int NCHUNK = KTOT / 32;  // 648
  __shared__ _Float16 __attribute__((aligned(32))) blds[2][512];

  const int nBase = blockIdx.x * 16;
  const int t = threadIdx.x;
  const int wave = t >> 5;
  const int lane = t & 31;
  const bool filler = (t < 64);  // waves 0,1 stage B

  // per-lane fill parameters: lane covers one contiguous 8-half run
  int fb = 0, fy = 0, fx = 0, fK = 0;
  unsigned ldsoff = 0;
  if (filler) {
    int L = t >> 1, run = t & 1;
    int col = L & 15;
    int khi = (L >= 16) ? 8 : 0;
    fK = run ? (16 + khi) : khi;
    int gn = nBase + col;          // grid is exact: gn always < 18432
    fb = gn / 36;
    int rem = gn % 36;
    fy = 2 * (rem / 6);
    fx = 2 * (rem % 6);
    ldsoff = (unsigned)(L * 16 + run * 8) * 2u;
  }
  const unsigned ldsbase = (unsigned)(size_t)(&blds[0][0]);

  auto issue = [&](int kc, int buf) {
    // K chunk = 32 consecutive ci at tap pix: pix = kc>>8, cibase = kc&255
    int pix = kc >> 8, cib = kc & 255;
    int ky = pix / 9, kx = pix % 9;
    const _Float16* gp =
        src + (((size_t)(fb * 20 + fy + ky) * 20 + (fx + kx)) << 8) + cib + fK;
    unsigned la = ldsbase + (unsigned)buf * 1024u + ldsoff;
    asm volatile("global_load_async_to_lds_b128 %0, %1, off"
                 :: "v"(la), "v"(gp) : "memory");
  };

  if (filler) issue(0, 0);

  v8f acc0 = {}; v8f acc1 = {};
  const int co0 = wave * 32;

  for (int kci = 0; kci < NCHUNK; ++kci) {
    const int kc = kci * 32;
    const int cur = kci & 1;
    if (filler && (kci + 1 < NCHUNK)) issue(kc + 32, cur ^ 1);
    if (kci + 1 < NCHUNK)
      asm volatile("s_wait_asynccnt 0x1" ::: "memory");  // prev chunk landed
    else
      asm volatile("s_wait_asynccnt 0x0" ::: "memory");
    __syncthreads();

    v16h bf = *(const v16h*)(&blds[cur][lane * 16]);
    v16h a0 = load_frag_rm(w, KTOT, co0 + (lane & 15), kc, lane);
    v16h a1 = load_frag_rm(w, KTOT, co0 + 16 + (lane & 15), kc, lane);
    acc0 = __builtin_amdgcn_wmma_f32_16x16x32_f16(false, a0, false, bf, (short)0, acc0, false, false);
    acc1 = __builtin_amdgcn_wmma_f32_16x16x32_f16(false, a1, false, bf, (short)0, acc1, false, false);
    __syncthreads();
  }

  int gn = nBase + (lane & 15);
  int b_ = gn / 36, sp = gn % 36;  // sp = oy*6+ox
  int rowOff = (lane >= 16) ? 8 : 0;
#pragma unroll
  for (int r = 0; r < 8; ++r) {
#pragma unroll
    for (int tt = 0; tt < 2; ++tt) {
      int co = co0 + tt * 16 + rowOff + r;
      float val = (tt == 0 ? acc0[r] : acc1[r]) + bias[co];
      // NHWC caps reshape: i = sp*32 + co/8, dim = co%8
      p[((size_t)b_ * NCAPS + sp * 32 + (co >> 3)) * 8 + (co & 7)] = val;
    }
  }
}

// ---- routing ---------------------------------------------------------------
__global__ void k_softmax_c(const float* __restrict__ bb, float* __restrict__ c, int total) {
  int g = blockIdx.x * blockDim.x + threadIdx.x;
  if (g >= total) return;
  const float* br = bb + (size_t)g * JCAPS;
  float m = br[0];
#pragma unroll
  for (int j = 1; j < JCAPS; ++j) m = fmaxf(m, br[j]);
  float e[JCAPS], s = 0.f;
#pragma unroll
  for (int j = 0; j < JCAPS; ++j) { e[j] = __expf(br[j] - m); s += e[j]; }
  float inv = 1.f / s;
#pragma unroll
  for (int j = 0; j < JCAPS; ++j) c[(size_t)g * JCAPS + j] = e[j] * inv;
}

// s[j,a] = sum_i c[n,i,j]*(sum_b W[i,j,a,b] p[n,i,b]); v = squash(s); |v| scores
__global__ __launch_bounds__(256) void k_routing_sv(const float* __restrict__ c,
                                                    const float* __restrict__ p,
                                                    const float* __restrict__ W,
                                                    float* __restrict__ v,
                                                    float* __restrict__ scores) {
  __shared__ float slds[JCAPS * DVEC];
  const int n = blockIdx.x, t = threadIdx.x;
  if (t < JCAPS * DVEC) {
    const int j = t / DVEC, a = t % DVEC;
    const float* pn = p + (size_t)n * NCAPS * 8;
    const float* cn = c + (size_t)n * NCAPS * JCAPS;
    float s = 0.f;
    for (int i = 0; i < NCAPS; ++i) {
      const float* wr = W + (((size_t)i * JCAPS + j) * DVEC + a) * 8;
      const float* pr = pn + (size_t)i * 8;
      float u = 0.f;
#pragma unroll
      for (int b = 0; b < 8; ++b) u += wr[b] * pr[b];
      s += cn[(size_t)i * JCAPS + j] * u;
    }
    slds[t] = s;
  }
  __syncthreads();
  if (t < JCAPS * DVEC) {
    const int j = t / DVEC;
    float n2 = 0.f;
#pragma unroll
    for (int a2 = 0; a2 < DVEC; ++a2) { float x = slds[j * DVEC + a2]; n2 += x * x; }
    float nn = sqrtf(n2);
    float f = (nn > 0.f) ? n2 / ((1.f + n2) * nn) : 0.f;
    v[(size_t)n * (JCAPS * DVEC) + t] = slds[t] * f;
    if ((t % DVEC) == 0) scores[n * JCAPS + j] = n2 / (1.f + n2);
  }
}

// b[n,i,j] += sum_a (sum_b W[i,j,a,b] p[n,i,b]) * v[n,j,a]
__global__ __launch_bounds__(256) void k_routing_bupd(float* __restrict__ bb,
                                                      const float* __restrict__ p,
                                                      const float* __restrict__ W,
                                                      const float* __restrict__ v) {
  __shared__ float vlds[JCAPS * DVEC];
  const int n = blockIdx.x, t = threadIdx.x;
  if (t < JCAPS * DVEC) vlds[t] = v[(size_t)n * (JCAPS * DVEC) + t];
  __syncthreads();
  for (int i = t; i < NCAPS; i += 256) {
    const float* pr = p + ((size_t)n * NCAPS + i) * 8;
    float pv[8];
#pragma unroll
    for (int b = 0; b < 8; ++b) pv[b] = pr[b];
    for (int j = 0; j < JCAPS; ++j) {
      const float* wr = W + ((size_t)i * JCAPS + j) * (DVEC * 8);
      float acc = 0.f;
#pragma unroll
      for (int a = 0; a < DVEC; ++a) {
        float u = 0.f;
#pragma unroll
        for (int b = 0; b < 8; ++b) u += wr[a * 8 + b] * pv[b];
        acc += u * vlds[j * DVEC + a];
      }
      bb[((size_t)n * NCAPS + i) * JCAPS + j] += acc;
    }
  }
}

// sel[n,k] = v[n, y[n], k], f16, K padded 16 -> 32
__global__ void k_select_caps(const float* __restrict__ v, const int* __restrict__ y,
                              _Float16* __restrict__ sel) {
  int idx = blockIdx.x * blockDim.x + threadIdx.x;
  if (idx >= B_SZ * 32) return;
  int n = idx / 32, k = idx % 32;
  sel[idx] = (k < 16) ? (_Float16)v[(size_t)n * (JCAPS * DVEC) + y[n] * DVEC + k]
                      : (_Float16)0.f;
}

// ---- decoder GEMM: C = act(A_f16[M,K] @ BT_f16[N,K]^T + bias[N]) ----------
template <int ACT>  // 0: relu -> f16 out; 1: sigmoid -> f32 out
__global__ __launch_bounds__(256) void k_gemm_wmma(const _Float16* __restrict__ A,
                                                   const _Float16* __restrict__ BT,
                                                   const float* __restrict__ bias,
                                                   void* __restrict__ out,
                                                   int M, int N, int K) {
  const int wave = threadIdx.x >> 5, lane = threadIdx.x & 31;
  const int tile = blockIdx.x * 8 + wave;
  const int ntiles = (M / 16) * (N / 16);
  if (tile >= ntiles) return;
  const int mT = tile / (N / 16), nT = tile % (N / 16);

  v8f acc = {};
  for (int kc = 0; kc < K; kc += 32) {
    v16h a = load_frag_rm(A, K, mT * 16 + (lane & 15), kc, lane);
    v16h b = load_frag_rm(BT, K, nT * 16 + (lane & 15), kc, lane);
    acc = __builtin_amdgcn_wmma_f32_16x16x32_f16(false, a, false, b, (short)0, acc, false, false);
  }
  const int col = nT * 16 + (lane & 15);
  const int rowOff = (lane >= 16) ? 8 : 0;
#pragma unroll
  for (int r = 0; r < 8; ++r) {
    int gm = mT * 16 + rowOff + r;
    float val = acc[r] + bias[col];
    if (ACT == 0) {
      val = val > 0.f ? val : 0.f;
      ((_Float16*)out)[(size_t)gm * N + col] = (_Float16)val;
    } else {
      val = 1.f / (1.f + __expf(-val));
      ((float*)out)[(size_t)gm * N + col] = val;
    }
  }
}

// ---------------------------------------------------------------------------
extern "C" void kernel_launch(void* const* d_in, const int* in_sizes, int n_in,
                              void* d_out, int out_size, void* d_ws, size_t ws_size,
                              hipStream_t stream) {
  (void)in_sizes; (void)n_in; (void)out_size; (void)ws_size;
  const float* x       = (const float*)d_in[0];
  const int*   y       = (const int*)  d_in[1];
  const float* conv1_w = (const float*)d_in[2];
  const float* conv1_b = (const float*)d_in[3];
  const float* pc_w    = (const float*)d_in[4];
  const float* pc_b    = (const float*)d_in[5];
  const float* W       = (const float*)d_in[6];
  const float* w1      = (const float*)d_in[7];
  const float* b1      = (const float*)d_in[8];
  const float* w2      = (const float*)d_in[9];
  const float* b2      = (const float*)d_in[10];
  const float* w3      = (const float*)d_in[11];
  const float* b3      = (const float*)d_in[12];
  float* out_scores = (float*)d_out;         // [512,10]
  float* out_recon  = (float*)d_out + 5120;  // [512,784]

  char* base = (char*)d_ws;
  size_t off = 0;
  auto alloc = [&](size_t bytes) -> void* {
    void* p = base + off;
    off = (off + bytes + 255) & ~(size_t)255;
    return p;
  };
  _Float16* x16   = (_Float16*)alloc((size_t)B_SZ * 784 * 2);
  _Float16* w1c   = (_Float16*)alloc((size_t)256 * 96 * 2);
  _Float16* h16   = (_Float16*)alloc((size_t)B_SZ * 400 * 256 * 2);     // NHWC, 105 MB
  _Float16* pcw16 = (_Float16*)alloc((size_t)256 * 20736 * 2);          // 10.6 MB
  float*    p32   = (float*)   alloc((size_t)B_SZ * NCAPS * 8 * 4);     // 19 MB
  float*    bbuf  = (float*)   alloc((size_t)B_SZ * NCAPS * JCAPS * 4); // 24 MB
  float*    cbuf  = (float*)   alloc((size_t)B_SZ * NCAPS * JCAPS * 4); // 24 MB
  float*    vbuf  = (float*)   alloc((size_t)B_SZ * JCAPS * DVEC * 4);
  _Float16* sel16 = (_Float16*)alloc((size_t)B_SZ * 32 * 2);
  _Float16* w1T   = (_Float16*)alloc((size_t)512 * 32 * 2);
  _Float16* w2T   = (_Float16*)alloc((size_t)1024 * 512 * 2);
  _Float16* w3T   = (_Float16*)alloc((size_t)784 * 1024 * 2);
  _Float16* d1    = (_Float16*)alloc((size_t)B_SZ * 512 * 2);
  _Float16* d2    = (_Float16*)alloc((size_t)B_SZ * 1024 * 2);

  const int TB = 256;
  auto cdiv = [](int a, int b) { return (a + b - 1) / b; };

  // --- prep / conversions ---
  k_cvt_f32_f16<<<cdiv(B_SZ * 784, TB), TB, 0, stream>>>(x, x16, B_SZ * 784);
  k_pack_conv1_w<<<cdiv(256 * 96, TB), TB, 0, stream>>>(conv1_w, w1c);
  k_pack_pcw<<<cdiv(256 * 20736, TB), TB, 0, stream>>>(pc_w, pcw16);
  k_transpose_pad<<<cdiv(512 * 32, TB), TB, 0, stream>>>(w1, w1T, 16, 32, 512);
  k_transpose_pad<<<cdiv(1024 * 512, TB), TB, 0, stream>>>(w2, w2T, 512, 512, 1024);
  k_transpose_pad<<<cdiv(784 * 1024, TB), TB, 0, stream>>>(w3, w3T, 1024, 1024, 784);
  k_fill_zero_f32<<<2048, TB, 0, stream>>>(bbuf, B_SZ * NCAPS * JCAPS);

  // --- conv1 (N = 512*400 = 204800) -> h NHWC f16 ---
  k_conv1_wmma<<<204800 / 16, TB, 0, stream>>>(w1c, x16, conv1_b, h16);
  // --- primary caps conv (196 GFLOP, async double-buffered B) ---
  k_pcconv_wmma<<<18432 / 16, TB, 0, stream>>>(pcw16, h16, pc_b, p32);

  // --- dynamic routing, 3 iterations (u recomputed on the fly) ---
  const int NI = B_SZ * NCAPS;  // 589824
  for (int it = 0; it < 3; ++it) {
    k_softmax_c<<<cdiv(NI, TB), TB, 0, stream>>>(bbuf, cbuf, NI);
    k_routing_sv<<<B_SZ, TB, 0, stream>>>(cbuf, p32, W, vbuf, out_scores);
    if (it < 2)
      k_routing_bupd<<<B_SZ, TB, 0, stream>>>(bbuf, p32, W, vbuf);
  }

  // --- decoder ---
  k_select_caps<<<cdiv(B_SZ * 32, TB), TB, 0, stream>>>(vbuf, y, sel16);
  k_gemm_wmma<0><<<128, TB, 0, stream>>>(sel16, w1T, b1, d1, 512, 512, 32);
  k_gemm_wmma<0><<<256, TB, 0, stream>>>(d1, w2T, b2, d2, 512, 1024, 512);
  k_gemm_wmma<1><<<196, TB, 0, stream>>>(d2, w3T, b3, out_recon, 512, 784, 1024);
}
